// CapsTimeModel_3856880632338
// MI455X (gfx1250) — matrix operand order repaired and checked by reading, not compiled, not run on hardware
//
#include <hip/hip_runtime.h>

// ---------------------------------------------------------------------------
// Types for CDNA5 WMMA
// ---------------------------------------------------------------------------
typedef __attribute__((ext_vector_type(16))) __bf16 v16bf;
typedef __attribute__((ext_vector_type(8)))  float  v8f;

__device__ inline unsigned short f2bf(float f) {
  unsigned int u = __float_as_uint(f);
  unsigned int r = (u + 0x7FFFu + ((u >> 16) & 1u)) >> 16;   // round-nearest-even
  return (unsigned short)r;
}

// Async global->LDS copy (CDNA5 ASYNCcnt path).
__device__ inline void async_b128(void* lds, const void* g) {
  unsigned loff = (unsigned)(unsigned long long)lds;   // low 32 bits = LDS offset
  asm volatile("global_load_async_to_lds_b128 %0, %1, off"
               :: "v"(loff), "v"(g) : "memory");
}
__device__ inline void async_wait_le3() { asm volatile("s_wait_asynccnt 3" ::: "memory"); }
__device__ inline void async_wait_le0() { asm volatile("s_wait_asynccnt 0" ::: "memory"); }

// ---------------------------------------------------------------------------
// K1: conv1 (3->128, 3x3, s2, p1) + bias + ReLU, output bf16 (64,128,32,32)
// ---------------------------------------------------------------------------
__global__ __launch_bounds__(256) void conv1_relu_kernel(
    const float* __restrict__ x, const float* __restrict__ w,
    const float* __restrict__ bias, unsigned short* __restrict__ out) {
  long long i = (long long)blockIdx.x * 256 + threadIdx.x;
  const long long total = 64LL * 128 * 32 * 32;
  if (i >= total) return;
  int ow = (int)(i & 31), oh = (int)((i >> 5) & 31);
  int oc = (int)((i >> 10) & 127), b = (int)(i >> 17);
  float acc = bias[oc];
  #pragma unroll
  for (int ic = 0; ic < 3; ++ic)
    #pragma unroll
    for (int kh = 0; kh < 3; ++kh) {
      int ih = oh * 2 - 1 + kh;
      if (ih < 0 || ih >= 64) continue;
      #pragma unroll
      for (int kw = 0; kw < 3; ++kw) {
        int iw = ow * 2 - 1 + kw;
        if (iw < 0 || iw >= 64) continue;
        acc += x[(((long long)b * 3 + ic) * 64 + ih) * 64 + iw] *
               w[((oc * 3 + ic) * 3 + kh) * 3 + kw];
      }
    }
  out[i] = f2bf(fmaxf(acc, 0.f));
}

// ---------------------------------------------------------------------------
// K2: generic f32 -> bf16 cast (used for pc_w: (512,1152) is already N x K)
// ---------------------------------------------------------------------------
__global__ __launch_bounds__(256) void cast_bf16_kernel(
    const float* __restrict__ in, unsigned short* __restrict__ out, long long n) {
  long long i = (long long)blockIdx.x * 256 + threadIdx.x;
  if (i < n) out[i] = f2bf(in[i]);
}

// ---------------------------------------------------------------------------
// K3: im2col for conv2: A2[m=(b,oh,ow)][k=(ic,kh,kw)] bf16, M=16384 K=1152
// ---------------------------------------------------------------------------
__global__ __launch_bounds__(256) void im2col2_kernel(
    const unsigned short* __restrict__ c, unsigned short* __restrict__ A) {
  long long i = (long long)blockIdx.x * 256 + threadIdx.x;
  const long long total = 16384LL * 1152;
  if (i >= total) return;
  int k = (int)(i % 1152);
  long long m = i / 1152;
  int ow = (int)(m & 15), oh = (int)((m >> 4) & 15), b = (int)(m >> 8);
  int kw = k % 3, r = k / 3, kh = r % 3, ic = r / 3;
  int ih = oh * 2 - 1 + kh, iw = ow * 2 - 1 + kw;
  unsigned short v = 0;
  if (ih >= 0 && ih < 32 && iw >= 0 && iw < 32)
    v = c[(((long long)b * 128 + ic) * 32 + ih) * 32 + iw];
  A[i] = v;
}

// ---------------------------------------------------------------------------
// K6: im2col for caps conv: A3[row=(b,h,w,a)][k=(n,kk,ll,x)] bf16 from val f32
// M=12544  K=1152
// ---------------------------------------------------------------------------
__global__ __launch_bounds__(256) void im2col3_kernel(
    const float* __restrict__ val, unsigned short* __restrict__ A) {
  long long i = (long long)blockIdx.x * 256 + threadIdx.x;
  const long long total = 12544LL * 1152;
  if (i >= total) return;
  int k = (int)(i % 1152);
  long long row = i / 1152;
  int a = (int)(row & 3);
  long long q = row >> 2;
  int w = (int)(q % 7); q /= 7;
  int h = (int)(q % 7);
  int b = (int)(q / 7);
  int x = k & 3; int q2 = k >> 2;
  int ll = q2 % 3; int p = q2 / 3;
  int kk = p % 3; int n = p / 3;
  float v = val[((((long long)b * 32 + n) * 16 + (2 * h + kk)) * 16 + (2 * w + ll)) * 16 + a * 4 + x];
  A[i] = f2bf(v);
}

// ---------------------------------------------------------------------------
// K7: pack caps_w (3,3,32,4,4,32) -> B3[c=(m*4+d)][k=(n,kk,ll,x)] bf16 (128x1152)
// ---------------------------------------------------------------------------
__global__ __launch_bounds__(256) void pack_w3_kernel(
    const float* __restrict__ cw, unsigned short* __restrict__ B) {
  int i = blockIdx.x * 256 + threadIdx.x;
  const int total = 128 * 1152;
  if (i >= total) return;
  int k = i % 1152, c = i / 1152;
  int mcap = c >> 2, dd = c & 3;
  int x = k & 3; int q2 = k >> 2;
  int ll = q2 % 3; int p = q2 / 3;
  int kk = p % 3; int n = p / 3;
  // caps_w[kk][ll][n][x][dd][mcap]
  long long idx = ((((long long)(kk * 3 + ll) * 32 + n) * 4 + x) * 4 + dd) * 32 + mcap;
  B[i] = f2bf(cw[idx]);
}

// ---------------------------------------------------------------------------
// K4/K8: bf16 WMMA GEMM  out = scale * A(MxK) * Bt(NxK)^T  with epilogue scatter
//   block tile 128(M) x 64(N), K-step 32; 8 waves, one 16x64 sub-tile per wave
//   (4 WMMAs per K-step sharing one A fragment; all fragments loaded up front
//    so ds_load latency overlaps the WMMA pipeline)
//   Double-buffered LDS: async loads for chunk i+1 are issued before computing
//   on chunk i; s_wait_asynccnt 3 keeps exactly the newest stage in flight
//   (every wave issues exactly 3 async b128 per stage: 2 for A, 1 for B).
//   mode 0: conv2 -> u_raw laid out as (b, ncaps, oh, ow, d)      [M=16384,N=512]
//   mode 1: caps  -> v_raw laid out as (b, mcap, h, w, a*4+d)/32  [M=12544,N=128]
// ---------------------------------------------------------------------------
#define GK_BM 128
#define GK_BN 64
#define GK_BK 32
#define GK_LD 40   // padded LDS row stride (bf16 elems); 80B keeps 16B alignment

__global__ __launch_bounds__(256) void gemm_bf16_wmma_kernel(
    const unsigned short* __restrict__ A, const unsigned short* __restrict__ Bt,
    float* __restrict__ out, int M, int N, int K, int mode, float scale) {
  __shared__ unsigned short As[2][GK_BM * GK_LD];   // 2 x 10.0 KB
  __shared__ unsigned short Bs[2][GK_BN * GK_LD];   // 2 x  5.0 KB

  const int t = threadIdx.x;
  const int wave = t >> 5, lane = t & 31;
  const int half = lane >> 4, l16 = lane & 15;

  const int nblk = N / GK_BN;
  const int m0 = ((int)blockIdx.x / nblk) * GK_BM;
  const int n0 = ((int)blockIdx.x % nblk) * GK_BN;

  v8f acc[4] = {{}, {}, {}, {}};

  // staging assignments: A chunk = 128 rows x 64B = 512 b128 (2 per thread),
  //                      B chunk =  64 rows x 64B = 256 b128 (1 per thread)
  const int ar0 = t >> 2,          as0 = t & 3;
  const int ar1 = (t + 256) >> 2,  as1 = (t + 256) & 3;
  const int br  = t >> 2,          bsg = t & 3;

  const unsigned short* Arow0 = A  + (long long)(m0 + ar0) * K + as0 * 8;
  const unsigned short* Arow1 = A  + (long long)(m0 + ar1) * K + as1 * 8;
  const unsigned short* Brow  = Bt + (long long)(n0 + br)  * K + bsg * 8;

  unsigned short* AsD0 = &As[0][ar0 * GK_LD + as0 * 8];
  unsigned short* AsD1 = &As[0][ar1 * GK_LD + as1 * 8];
  unsigned short* BsD  = &Bs[0][br  * GK_LD + bsg * 8];
  const int AsStride = GK_BM * GK_LD;   // elems between buffers
  const int BsStride = GK_BN * GK_LD;

  // prologue: stage buffer 0 with chunk 0
  async_b128(AsD0, Arow0);
  async_b128(AsD1, Arow1);
  async_b128(BsD,  Brow);

  const int nIter = K / GK_BK;
  for (int it = 0; it < nIter; ++it) {
    const int p = it & 1;
    if (it + 1 < nIter) {
      const int q = p ^ 1;
      const int kn = (it + 1) * GK_BK;
      async_b128(AsD0 + q * AsStride, Arow0 + kn);
      async_b128(AsD1 + q * AsStride, Arow1 + kn);
      async_b128(BsD  + q * BsStride, Brow  + kn);
      __builtin_prefetch(Arow0 + kn + GK_BK, 0, 1);   // global_prefetch_b8
      async_wait_le3();   // wait for chunk `it` only; keep newest stage in flight
    } else {
      async_wait_le0();
    }
    __syncthreads();

    // Load ALL fragments first (10x ds_load_b128) so the compiler can overlap
    // LDS return latency with the WMMA pipeline (descending s_wait_dscnt).
    const int mrow = wave * 16 + l16;
    union { uint4 u4[2]; v16bf v; } af, bfr0, bfr1, bfr2, bfr3;
    // A fragment: lane holds row m=l16, 16 contiguous k starting at half*16
    af.u4[0] = *(const uint4*)&As[p][mrow * GK_LD + half * 16];
    af.u4[1] = *(const uint4*)&As[p][mrow * GK_LD + half * 16 + 8];
    // B fragments: lane holds col n, 16 contiguous k starting at half*16
    const unsigned short* bp0 = &Bs[p][(0 * 16 + l16) * GK_LD + half * 16];
    const unsigned short* bp1 = &Bs[p][(1 * 16 + l16) * GK_LD + half * 16];
    const unsigned short* bp2 = &Bs[p][(2 * 16 + l16) * GK_LD + half * 16];
    const unsigned short* bp3 = &Bs[p][(3 * 16 + l16) * GK_LD + half * 16];
    bfr0.u4[0] = *(const uint4*)bp0;  bfr0.u4[1] = *(const uint4*)(bp0 + 8);
    bfr1.u4[0] = *(const uint4*)bp1;  bfr1.u4[1] = *(const uint4*)(bp1 + 8);
    bfr2.u4[0] = *(const uint4*)bp2;  bfr2.u4[1] = *(const uint4*)(bp2 + 8);
    bfr3.u4[0] = *(const uint4*)bp3;  bfr3.u4[1] = *(const uint4*)(bp3 + 8);

    acc[0] = __builtin_amdgcn_wmma_f32_16x16x32_bf16(false, af.v, false, bfr0.v,
                                                     (short)0, acc[0], false, false);
    acc[1] = __builtin_amdgcn_wmma_f32_16x16x32_bf16(false, af.v, false, bfr1.v,
                                                     (short)0, acc[1], false, false);
    acc[2] = __builtin_amdgcn_wmma_f32_16x16x32_bf16(false, af.v, false, bfr2.v,
                                                     (short)0, acc[2], false, false);
    acc[3] = __builtin_amdgcn_wmma_f32_16x16x32_bf16(false, af.v, false, bfr3.v,
                                                     (short)0, acc[3], false, false);
    __syncthreads();   // all waves done reading buffer p before it is re-staged
  }

  // epilogue: lane VGPR v holds (m = half*8 + v, n = l16) of each 16x16 tile
  #pragma unroll
  for (int v = 0; v < 8; ++v) {
    int m = m0 + wave * 16 + half * 8 + v;
    #pragma unroll
    for (int nt = 0; nt < 4; ++nt) {
      int n = n0 + nt * 16 + l16;
      float r = acc[nt][v] * scale;
      long long idx;
      if (mode == 0) {
        int b = m >> 8, oh = (m >> 4) & 15, ow = m & 15;
        int nc = n >> 4, d = n & 15;
        idx = ((((long long)b * 32 + nc) * 16 + oh) * 16 + ow) * 16 + d;
      } else {
        int a = m & 3; int q = m >> 2;
        int w = q % 7; q /= 7;
        int h = q % 7; int b = q / 7;
        int mc = n >> 2, dd = n & 3;
        idx = ((((long long)b * 32 + mc) * 7 + h) * 7 + w) * 16 + a * 4 + dd;
      }
      out[idx] = r;
    }
  }
}

// ---------------------------------------------------------------------------
// LayerNorm over last dim 16
// ---------------------------------------------------------------------------
__global__ __launch_bounds__(256) void ln16_kernel(
    const float* __restrict__ in, float* __restrict__ out,
    const float* __restrict__ g, const float* __restrict__ b, int rows) {
  int r = blockIdx.x * 256 + threadIdx.x;
  if (r >= rows) return;
  const float* p = in + (long long)r * 16;
  float mu = 0.f;
  #pragma unroll
  for (int i = 0; i < 16; ++i) mu += p[i];
  mu *= (1.f / 16.f);
  float var = 0.f;
  #pragma unroll
  for (int i = 0; i < 16; ++i) { float d = p[i] - mu; var += d * d; }
  var *= (1.f / 16.f);
  float inv = rsqrtf(var + 1e-5f);
  float* o = out + (long long)r * 16;
  #pragma unroll
  for (int i = 0; i < 16; ++i) o[i] = (p[i] - mu) * inv * g[i] + b[i];
}

// ---------------------------------------------------------------------------
// K10: fused routing pass for caps conv (qk -> softmax(m) -> weighted sum)
//   one block per (b,h,w) position; LDS < 64KB
// ---------------------------------------------------------------------------
__global__ __launch_bounds__(256) void caps_route_conv_kernel(
    const float* __restrict__ val, const float* __restrict__ vprev,
    const float* __restrict__ cw, float* __restrict__ outv) {
  __shared__ float u_s[288 * 16];     // 18.0 KB
  __shared__ float nv_s[32 * 16];     //  2.0 KB
  __shared__ float qk_s[288 * 32];    // 36.0 KB
  __shared__ float out_s[32 * 16];    //  2.0 KB

  const int t = threadIdx.x;
  const int bid = blockIdx.x;
  const int b = bid / 49, pos = bid % 49, h = pos / 7, w = pos % 7;

  for (int i = t; i < 288 * 16; i += 256) {
    int d = i & 15, nkl = i >> 4;
    int ll = nkl % 3, p2 = nkl / 3, kk = p2 % 3, n = p2 / 3;
    u_s[i] = val[((((long long)b * 32 + n) * 16 + (2 * h + kk)) * 16 + (2 * w + ll)) * 16 + d];
  }
  for (int i = t; i < 512; i += 256) {
    int d = i & 15, m = i >> 4;
    nv_s[i] = vprev[((((long long)b * 32 + m) * 7 + h) * 7 + w) * 16 + d];
    out_s[i] = 0.f;
  }
  __syncthreads();

  const float scale = 0.25f;   // 1/sqrt(16)
  // qk[nkl][m]
  for (int it = 0; it < 36; ++it) {
    int p = t + it * 256;
    int m = p & 31, nkl = p >> 5;
    int ll = nkl % 3, p2 = nkl / 3, kk = p2 % 3, n = p2 / 3;
    const float* u = &u_s[nkl * 16];
    const float* wb = cw + ((long long)((kk * 3 + ll) * 32 + n) * 16) * 32 + m;
    float acc = 0.f;
    #pragma unroll
    for (int a = 0; a < 4; ++a)
      #pragma unroll
      for (int d = 0; d < 4; ++d) {
        float pp = 0.f;
        #pragma unroll
        for (int x = 0; x < 4; ++x) pp += u[a * 4 + x] * wb[(x * 4 + d) * 32];
        acc += pp * nv_s[m * 16 + a * 4 + d];
      }
    qk_s[nkl * 32 + m] = acc * scale;
  }
  __syncthreads();

  // softmax over m (32)
  for (int r = t; r < 288; r += 256) {
    float* q = &qk_s[r * 32];
    float mx = q[0];
    for (int m = 1; m < 32; ++m) mx = fmaxf(mx, q[m]);
    float s = 0.f;
    for (int m = 0; m < 32; ++m) { float e = __expf(q[m] - mx); q[m] = e; s += e; }
    float inv = 1.f / s;
    for (int m = 0; m < 32; ++m) q[m] *= inv;
  }
  __syncthreads();

  // weighted accumulation: thread (grp, m) handles 36 nkl values
  {
    int m = t & 31, grp = t >> 5;
    float acc[16];
    #pragma unroll
    for (int i = 0; i < 16; ++i) acc[i] = 0.f;
    for (int nkl = grp * 36; nkl < grp * 36 + 36; ++nkl) {
      int ll = nkl % 3, p2 = nkl / 3, kk = p2 % 3, n = p2 / 3;
      const float* u = &u_s[nkl * 16];
      const float* wb = cw + ((long long)((kk * 3 + ll) * 32 + n) * 16) * 32 + m;
      float q = qk_s[nkl * 32 + m];
      #pragma unroll
      for (int a = 0; a < 4; ++a)
        #pragma unroll
        for (int d = 0; d < 4; ++d) {
          float pp = 0.f;
          #pragma unroll
          for (int x = 0; x < 4; ++x) pp += u[a * 4 + x] * wb[(x * 4 + d) * 32];
          acc[a * 4 + d] += q * pp;
        }
    }
    #pragma unroll
    for (int i = 0; i < 16; ++i) atomicAdd(&out_s[m * 16 + i], acc[i]);  // ds_add_f32
  }
  __syncthreads();

  for (int i = t; i < 512; i += 256) {
    int d = i & 15, m = i >> 4;
    outv[((((long long)b * 32 + m) * 7 + h) * 7 + w) * 16 + d] = out_s[i];
  }
}

// ---------------------------------------------------------------------------
// K12: FC caps pass 1 (no routing): p_raw[b][mc][a*4+d] = (1/10) sum_{j,x} ...
// ---------------------------------------------------------------------------
__global__ __launch_bounds__(256) void caps_fc1_kernel(
    const float* __restrict__ v2, const float* __restrict__ fw,
    float* __restrict__ praw) {
  int b = blockIdx.x, t = threadIdx.x;
  if (t >= 160) return;
  int mc = t / 16, dd = t % 16, a = dd >> 2, d = dd & 3;
  const float* u = v2 + (long long)b * 25088;   // (j,16) rows, j = m*49+h*7+w
  float acc = 0.f;
  for (int j = 0; j < 1568; ++j) {
    const float* ur = u + j * 16 + a * 4;
    const float* wr = fw + ((long long)j * 4) * 40 + d * 10 + mc;  // fw[j][x][d][mc]
    #pragma unroll
    for (int x = 0; x < 4; ++x) acc += ur[x] * wr[x * 40];
  }
  praw[(long long)b * 160 + mc * 16 + dd] = acc * 0.1f;
}

// ---------------------------------------------------------------------------
// K14: FC routing pass
// ---------------------------------------------------------------------------
__global__ __launch_bounds__(256) void caps_fc_route_kernel(
    const float* __restrict__ v2, const float* __restrict__ fw,
    const float* __restrict__ pprev, float* __restrict__ praw2) {
  __shared__ float nv_s[160];
  __shared__ float out_s[160];
  int b = blockIdx.x, t = threadIdx.x;
  if (t < 160) { nv_s[t] = pprev[(long long)b * 160 + t]; out_s[t] = 0.f; }
  __syncthreads();
  const float scale = 0.25f;
  const float* u = v2 + (long long)b * 25088;
  for (int j = t; j < 1568; j += 256) {
    float uj[16];
    #pragma unroll
    for (int i = 0; i < 16; ++i) uj[i] = u[j * 16 + i];
    const float* wj = fw + (long long)j * 160;   // [x][d][mc]
    float qk[10];
    float mx = -1e30f;
    for (int mc = 0; mc < 10; ++mc) {
      float acc = 0.f;
      #pragma unroll
      for (int a = 0; a < 4; ++a)
        #pragma unroll
        for (int d = 0; d < 4; ++d) {
          float pp = 0.f;
          #pragma unroll
          for (int x = 0; x < 4; ++x) pp += uj[a * 4 + x] * wj[(x * 4 + d) * 10 + mc];
          acc += pp * nv_s[mc * 16 + a * 4 + d];
        }
      qk[mc] = acc * scale;
      mx = fmaxf(mx, qk[mc]);
    }
    float s = 0.f;
    for (int mc = 0; mc < 10; ++mc) { qk[mc] = __expf(qk[mc] - mx); s += qk[mc]; }
    float inv = 1.f / s;
    for (int mc = 0; mc < 10; ++mc) {
      float q = qk[mc] * inv;
      #pragma unroll
      for (int a = 0; a < 4; ++a)
        #pragma unroll
        for (int d = 0; d < 4; ++d) {
          float pp = 0.f;
          #pragma unroll
          for (int x = 0; x < 4; ++x) pp += uj[a * 4 + x] * wj[(x * 4 + d) * 10 + mc];
          atomicAdd(&out_s[mc * 16 + a * 4 + d], q * pp);
        }
    }
  }
  __syncthreads();
  if (t < 160) praw2[(long long)b * 160 + t] = out_s[t];
}

// ---------------------------------------------------------------------------
// Host launcher
// ---------------------------------------------------------------------------
extern "C" void kernel_launch(void* const* d_in, const int* in_sizes, int n_in,
                              void* d_out, int out_size, void* d_ws, size_t ws_size,
                              hipStream_t stream) {
  const float* x     = (const float*)d_in[0];
  const float* bb_w  = (const float*)d_in[1];
  const float* bb_b  = (const float*)d_in[2];
  const float* pc_w  = (const float*)d_in[3];
  const float* ln0_g = (const float*)d_in[4];
  const float* ln0_b = (const float*)d_in[5];
  const float* capsw = (const float*)d_in[6];
  const float* ln1_g = (const float*)d_in[7];
  const float* ln1_b = (const float*)d_in[8];
  const float* fc_w  = (const float*)d_in[9];
  const float* ln2_g = (const float*)d_in[10];
  const float* ln2_b = (const float*)d_in[11];

  char* ws = (char*)d_ws;
  size_t off = 0;
  auto alloc = [&](size_t bytes) { void* p = ws + off; off = (off + bytes + 255) & ~(size_t)255; return p; };

  unsigned short* c_bf  = (unsigned short*)alloc(16777216);    // conv1 out bf16
  unsigned short* w2_bf = (unsigned short*)alloc(1179648);     // pc_w bf16 (512x1152)
  unsigned short* A2    = (unsigned short*)alloc(37748736);    // im2col conv2
  float* u_raw = (float*)alloc(8388608);                       // GEMM1 out (val layout)
  float* valb  = (float*)alloc(8388608);                       // LN0 out
  unsigned short* A3 = (unsigned short*)alloc(28901376);       // im2col caps
  unsigned short* B3 = (unsigned short*)alloc(294912);         // caps_w packed bf16
  float* v_raw  = (float*)alloc(6422528);
  float* v1     = (float*)alloc(6422528);
  float* v_raw2 = (float*)alloc(6422528);
  float* v2     = (float*)alloc(6422528);
  float* p_raw  = (float*)alloc(40960);
  float* p1     = (float*)alloc(40960);
  float* p_raw2 = (float*)alloc(40960);
  (void)ws_size; (void)n_in; (void)in_sizes; (void)out_size;

  // 1) conv1 + relu -> bf16
  conv1_relu_kernel<<<(64 * 128 * 1024 + 255) / 256, 256, 0, stream>>>(x, bb_w, bb_b, c_bf);
  // 2) pc_w -> bf16 (already N x K layout)
  cast_bf16_kernel<<<(589824 + 255) / 256, 256, 0, stream>>>(pc_w, w2_bf, 589824LL);
  // 3) im2col for conv2
  im2col2_kernel<<<(int)((16384LL * 1152 + 255) / 256), 256, 0, stream>>>(c_bf, A2);
  // 4) conv2 as WMMA GEMM: M=16384 N=512 K=1152 -> u_raw (val layout)
  gemm_bf16_wmma_kernel<<<(16384 / GK_BM) * (512 / GK_BN), 256, 0, stream>>>(
      A2, w2_bf, u_raw, 16384, 512, 1152, 0, 1.0f);
  // 5) LN0
  ln16_kernel<<<(524288 + 255) / 256, 256, 0, stream>>>(u_raw, valb, ln0_g, ln0_b, 524288);
  // 6) im2col for caps conv
  im2col3_kernel<<<(int)((12544LL * 1152 + 255) / 256), 256, 0, stream>>>(valb, A3);
  // 7) pack caps weights
  pack_w3_kernel<<<(147456 + 255) / 256, 256, 0, stream>>>(capsw, B3);
  // 8) caps conv pass 1 as WMMA GEMM: M=12544 N=128 K=1152, scale 1/32
  gemm_bf16_wmma_kernel<<<(12544 / GK_BM) * (128 / GK_BN), 256, 0, stream>>>(
      A3, B3, v_raw, 12544, 128, 1152, 1, 1.0f / 32.0f);
  // 9) LN1
  ln16_kernel<<<(100352 + 255) / 256, 256, 0, stream>>>(v_raw, v1, ln1_g, ln1_b, 100352);
  // 10) routing pass (qk -> softmax -> weighted sum)
  caps_route_conv_kernel<<<64 * 49, 256, 0, stream>>>(valb, v1, capsw, v_raw2);
  // 11) LN1 again
  ln16_kernel<<<(100352 + 255) / 256, 256, 0, stream>>>(v_raw2, v2, ln1_g, ln1_b, 100352);
  // 12) FC caps pass 1
  caps_fc1_kernel<<<64, 256, 0, stream>>>(v2, fc_w, p_raw);
  // 13) LN2
  ln16_kernel<<<(640 + 255) / 256, 256, 0, stream>>>(p_raw, p1, ln2_g, ln2_b, 640);
  // 14) FC routing pass
  caps_fc_route_kernel<<<64, 256, 0, stream>>>(v2, fc_w, p1, p_raw2);
  // 15) LN2 -> final output
  ln16_kernel<<<(640 + 255) / 256, 256, 0, stream>>>(p_raw2, (float*)d_out, ln2_g, ln2_b, 640);
}
